// AudioBridgingModule_11622181503389
// MI455X (gfx1250) — compile-verified, matrix operand order checked by reading, not run on hardware
//
#include <hip/hip_runtime.h>
#include <hip/hip_bf16.h>

// ---------------------------------------------------------------------------
// Problem constants (reference): B=2, TQ=4096, S=1024, D=256, H=8, DK=32, L=4
// rel_shift uses 2S = 2048 (power of two) -> pure index remap.
// ---------------------------------------------------------------------------
#define CB   2
#define CTQ  4096
#define CS   1024
#define CD   256
#define CH   8
#define CDK  32
#define CL   4
#define N2   2047            // 2S-1

typedef __attribute__((ext_vector_type(16))) _Float16 v16h;
typedef __attribute__((ext_vector_type(8)))  float    v8f;

// Async global->LDS copy (CDNA5). Guarded so compile never breaks if absent.
#if defined(__has_builtin)
#  if __has_builtin(__builtin_amdgcn_global_load_async_to_lds_b128) && \
      __has_builtin(__builtin_amdgcn_s_wait_asynccnt)
#    define HAS_ASYNC_LDS 1
#  endif
#endif
#ifndef HAS_ASYNC_LDS
#  define HAS_ASYNC_LDS 0
#endif

#if HAS_ASYNC_LDS
typedef int a_v4i __attribute__((vector_size(4 * sizeof(int))));
__device__ __forceinline__ void async_copy_b128(const void* g, void* l) {
  // generic->AS1 / generic->AS3 via integer reinterpret (LDS offset = addr[31:0])
  __builtin_amdgcn_global_load_async_to_lds_b128(
      (__attribute__((address_space(1))) a_v4i*)(unsigned long long)(uintptr_t)g,
      (__attribute__((address_space(3))) a_v4i*)(unsigned int)(uintptr_t)l,
      0, 0);
}
#endif

__device__ __forceinline__ v8f wmma_f32(v16h a, v16h b, v8f c) {
  return __builtin_amdgcn_wmma_f32_16x16x32_f16(false, a, false, b, (short)0, c,
                                                false, false);
}

// A-operand layout (CDNA5 ISA 7.12.2, 16-bit A 16x32):
//   lanes 0-15 : row = lane,     K in {0..7, 16..23}
//   lanes 16-31: row = lane-16,  K in {8..15, 24..31}
// B-operand assumed mirrored (treat B^T rows like A rows).
__device__ __forceinline__ v16h frag_lds_h(const _Float16* base, int stride, int lane) {
  int r  = lane & 15;
  int kb = (lane & 16) ? 8 : 0;
  const _Float16* p = base + r * stride + kb;
  v16h f;
#pragma unroll
  for (int j = 0; j < 8; ++j) { f[j] = p[j]; f[j + 8] = p[j + 16]; }
  return f;
}

__device__ __forceinline__ v16h frag_lds_f(const float* base, int stride, int lane) {
  int r  = lane & 15;
  int kb = (lane & 16) ? 8 : 0;
  const float* p = base + r * stride + kb;
  v16h f;
#pragma unroll
  for (int j = 0; j < 8; ++j) { f[j] = (_Float16)p[j]; f[j + 8] = (_Float16)p[j + 16]; }
  return f;
}

// element(n,k) = base[n*stride + k], f16 global rows -> B operand (b128 loads)
__device__ __forceinline__ v16h frag_g_rows_h(const _Float16* __restrict__ base,
                                              int stride, int lane) {
  int r  = lane & 15;
  int kb = (lane & 16) ? 8 : 0;
  const _Float16* p = base + (size_t)r * stride + kb;
  v16h f;
#pragma unroll
  for (int j = 0; j < 8; ++j) { f[j] = p[j]; f[j + 8] = p[j + 16]; }
  return f;
}

// ---------------------------------------------------------------------------
// Positional encoding pe(2S-1, D)
// ---------------------------------------------------------------------------
__global__ __launch_bounds__(128) void pe_kernel(float* __restrict__ pe) {
  int r = blockIdx.x;          // 0..2046
  int j = threadIdx.x;         // 0..127 (sin/cos pairs)
  float div = __expf((float)(2 * j) * (-9.210340371976184f / (float)CD));
  float pos = (r < CS) ? (float)(CS - 1 - r) : -(float)(r - (CS - 1));
  float arg = pos * div;
  pe[(size_t)r * CD + 2 * j]     = sinf(arg);
  pe[(size_t)r * CD + 2 * j + 1] = cosf(arg);
}

// ---------------------------------------------------------------------------
// C = A(MxK) @ W^T(NxK) + bias. f32 inputs, OT output (optionally transposed).
// Block = 128 threads (4 waves), tile 64x64, K stepped by 32 through LDS f16.
// ---------------------------------------------------------------------------
template <typename OT, bool TR>
__global__ __launch_bounds__(128) void gemm64(const float* __restrict__ A,
                                              const float* __restrict__ W,
                                              const float* __restrict__ bias,
                                              OT* __restrict__ C,
                                              int M, int N, int K, int ldc) {
  __shared__ _Float16 s_a[64 * 40];
  __shared__ _Float16 s_w[64 * 40];
  int m0 = blockIdx.x * 64, n0 = blockIdx.y * 64;
  int tid = threadIdx.x, lane = tid & 31, wave = tid >> 5;
  v8f acc[4] = {};

  for (int k0 = 0; k0 < K; k0 += 32) {
    for (int i = tid; i < 64 * 32; i += 128) {
      int r = i >> 5, c = i & 31;
      int ra = m0 + r; if (ra >= M) ra = M - 1;
      int rw = n0 + r; if (rw >= N) rw = N - 1;
      s_a[r * 40 + c] = (_Float16)A[(size_t)ra * K + k0 + c];
      s_w[r * 40 + c] = (_Float16)W[(size_t)rw * K + k0 + c];
      if (k0 + 32 < K) {  // global_prefetch_b8 hint for next k-tile
        __builtin_prefetch(&A[(size_t)ra * K + k0 + 32 + c], 0, 1);
      }
    }
    __syncthreads();
    v16h a = frag_lds_h(s_a + wave * 16 * 40, 40, lane);
#pragma unroll
    for (int nt = 0; nt < 4; ++nt) {
      v16h b = frag_lds_h(s_w + nt * 16 * 40, 40, lane);
      acc[nt] = wmma_f32(a, b, acc[nt]);
    }
    __syncthreads();
  }

  int hi = (lane & 16) ? 8 : 0;
  int cl = lane & 15;
#pragma unroll
  for (int nt = 0; nt < 4; ++nt) {
    int col = n0 + nt * 16 + cl;
    float bv = bias ? bias[col] : 0.0f;
#pragma unroll
    for (int j = 0; j < 8; ++j) {
      int row = m0 + wave * 16 + j + hi;
      if (row < M) {
        float val = acc[nt][j] + bv;
        if (TR) C[(size_t)col * ldc + row] = (OT)val;
        else    C[(size_t)row * ldc + col] = (OT)val;
      }
    }
  }
}

// ---------------------------------------------------------------------------
// Fused attention: one block per (b, h, 32-query tile). 128 threads = 4 waves.
// Scores row-resident in LDS -> exact softmax; rel-shift as plain WMMA GEMM
// (G = q_v . p^T, 48x176 tile) + skewed LDS gather:
//   bd[q,c] = G[q + 2 + ((c-q)>>11), ((c-q)&2047)-1], zero when (c-q)&2047==0
// V (transposed, f16) for this head is prefetched to LDS asynchronously at
// kernel start and drained with s_wait_asynccnt just before the attn.V WMMAs.
// ---------------------------------------------------------------------------
#define SCS 1032            // scores LDS stride (1024 + 8)
#define GS  180             // G LDS stride (176 + 4)
#define VTS 1032            // V^T LDS stride (1024 + 8 halfs)

__global__ __launch_bounds__(128) void attn_kernel(
    const _Float16* __restrict__ Q,    // (B,TQ,D) f16
    const _Float16* __restrict__ Km,   // (S,D)    f16
    const _Float16* __restrict__ Vt,   // (D,S)    f16 (transposed V)
    const _Float16* __restrict__ P,    // (2S-1,D) f16
    const float* __restrict__ pbu, const float* __restrict__ pbv,
    const unsigned char* __restrict__ mask,
    float* __restrict__ attn_out,      // d_out + B*TQ*D, layout (L,B,H,TQ,S)
    float* __restrict__ ctx,           // (B,TQ,D) f32
    int layer) {
  extern __shared__ char smem[];
  float*    s_sc  = (float*)smem;                 // 32*SCS f32
  float*    s_g   = s_sc + 32 * SCS;              // 48*GS  f32
  float*    s_red = s_g + 48 * GS;                // 32*4   f32
  _Float16* s_qu  = (_Float16*)(s_red + 32 * 4);  // 32*40
  _Float16* s_qv  = s_qu + 32 * 40;               // 48*40
  _Float16* s_pb  = s_qv + 48 * 40;               // 176*40
  _Float16* s_vt  = s_pb + 176 * 40;              // 32*VTS

  int bid = blockIdx.x;
  int qb = bid & 127;
  int h  = (bid >> 7) & 7;
  int b  = bid >> 10;
  int q0 = qb * 32;

  int tid = threadIdx.x, lane = tid & 31, wave = tid >> 5;

  // ---- kick off async prefetch of V^T head rows into LDS (64 KB) ----------
  {
    const _Float16* vsrc = Vt + (size_t)(h * CDK) * CS;
#if HAS_ASYNC_LDS
    for (int i = tid; i < 32 * 128; i += 128) {
      int r = i >> 7, ch = i & 127;
      async_copy_b128(vsrc + (size_t)r * CS + ch * 8, s_vt + r * VTS + ch * 8);
    }
#else
    for (int i = tid; i < 32 * 128; i += 128) {
      int r = i >> 7, ch = i & 127;
      *(uint4*)(s_vt + r * VTS + ch * 8) =
          *(const uint4*)(vsrc + (size_t)r * CS + ch * 8);
    }
#endif
  }

  const _Float16* Qb = Q + (size_t)b * CTQ * CD;

  // ---- stage q_u (32x32) and q_v (48x32: rows q0..q0+47 for shifted rows) --
  for (int i = tid; i < 48 * 32; i += 128) {
    int r = i >> 5, d = i & 31;
    int qr = q0 + r; if (qr >= CTQ) qr = CTQ - 1;
    float qval = (float)Qb[(size_t)qr * CD + h * CDK + d];
    s_qv[r * 40 + d] = (_Float16)(qval + pbv[h * CDK + d]);
    if (r < 32) s_qu[r * 40 + d] = (_Float16)(qval + pbu[h * CDK + d]);
  }
  __syncthreads();

  // ---- ac = q_u . k^T : wave = (q-subtile, key-half); 32 WMMAs per wave ----
  {
    int sub = wave & 1, kh = wave >> 1;
    v16h a = frag_lds_h(s_qu + sub * 16 * 40, 40, lane);
    int hi = (lane & 16) ? 8 : 0, cl = lane & 15;
    for (int kt = 0; kt < 32; ++kt) {
      int s0 = kh * 512 + kt * 16;
      v16h bf = frag_g_rows_h(Km + (size_t)s0 * CD + h * CDK, CD, lane);
      v8f c = {};
      c = wmma_f32(a, bf, c);
#pragma unroll
      for (int j = 0; j < 8; ++j)
        s_sc[(sub * 16 + j + hi) * SCS + s0 + cl] = c[j];
    }
  }
  __syncthreads();

  // ---- bd via G-GEMM + skewed gather, 128-key chunks -----------------------
  for (int c0 = 0; c0 < CS; c0 += 128) {
    // stage 176 wrapped p-rows as f16 (16B chunks); zero row when col==0
    for (int i = tid; i < 176 * 4; i += 128) {
      int mrow = i >> 2, ch = i & 3;
      int v = c0 - q0 - 33 + mrow;        // equals (c - q) for gathered elems
      int col = v & 2047;
      uint4 val = make_uint4(0u, 0u, 0u, 0u);
      if (col != 0) {
        int jp = col - 1;                 // 0..2046
        val = *(const uint4*)(P + (size_t)jp * CD + h * CDK + ch * 8);
      }
      *(uint4*)(s_pb + mrow * 40 + ch * 8) = val;
    }
    __syncthreads();
    // G[48 x 176] = q_v(48x32) . p_sel^T : 3 Mtiles x 11 Ntiles = 33 WMMAs
    for (int t = wave; t < 33; t += 4) {
      int mt = t / 11, nt = t % 11;
      v16h a  = frag_lds_h(s_qv + mt * 16 * 40, 40, lane);
      v16h bf = frag_lds_h(s_pb + nt * 16 * 40, 40, lane);
      v8f c = {};
      c = wmma_f32(a, bf, c);
      int hi = (lane & 16) ? 8 : 0, cl = lane & 15;
#pragma unroll
      for (int j = 0; j < 8; ++j)
        s_g[(mt * 16 + j + hi) * GS + nt * 16 + cl] = c[j];
    }
    __syncthreads();
    // gather: scores[q][c] += G[q + 2 + ((c-q)>>11) - q0][(c-c0)-(q-q0)+33]
    for (int i = tid; i < 32 * 128; i += 128) {
      int ql = i >> 7, cc = i & 127;
      int q = q0 + ql, c = c0 + cc;
      int t2 = c - q;
      int r = ql + 2 + (t2 >> 11);        // arithmetic shift: {0,-1,-2}
      int m = cc - ql + 33;               // 2..160 < 176
      s_sc[ql * SCS + c] += s_g[r * GS + m];
    }
    __syncthreads();
  }

  // ---- scale + mask + softmax (4 threads per row) + write attn weights ----
  {
    const float scale = 0.17677669529663687f;   // 1/sqrt(32)
    const unsigned char* mk = mask + (size_t)b * CS;
    int row = tid >> 2, t4 = tid & 3;
    float* sr = s_sc + row * SCS;
    float mx = -3.0e38f;
    for (int c = t4; c < CS; c += 4) {
      float v = sr[c] * scale;
      if (!mk[c]) v = -3.0e38f;
      sr[c] = v;
      mx = fmaxf(mx, v);
    }
    s_red[row * 4 + t4] = mx;
    __syncthreads();
    mx = fmaxf(fmaxf(s_red[row * 4 + 0], s_red[row * 4 + 1]),
               fmaxf(s_red[row * 4 + 2], s_red[row * 4 + 3]));
    float sm = 0.0f;
    for (int c = t4; c < CS; c += 4) {
      float e = mk[c] ? __expf(sr[c] - mx) : 0.0f;
      sr[c] = e;
      sm += e;
    }
    __syncthreads();
    s_red[row * 4 + t4] = sm;
    __syncthreads();
    float tot = s_red[row * 4 + 0] + s_red[row * 4 + 1] +
                s_red[row * 4 + 2] + s_red[row * 4 + 3];
    float inv = (tot > 0.0f) ? (1.0f / tot) : 0.0f;
    size_t ob = ((((size_t)layer * CB + b) * CH + h) * CTQ + (q0 + row)) * CS;
    for (int c = t4; c < CS; c += 4) {
      float a = sr[c] * inv;
      sr[c] = a;                          // keep normalized attn for PV
      attn_out[ob + c] = a;
    }
  }

  // ---- drain async V^T prefetch, then ctx = attn . V ----------------------
#if HAS_ASYNC_LDS
  __builtin_amdgcn_s_wait_asynccnt(0);
#endif
  __syncthreads();
  {
    int sub = wave & 1, nh = wave >> 1, n0c = nh * 16;
    v8f acc = {};
    for (int kc = 0; kc < 32; ++kc) {
      v16h a  = frag_lds_f(s_sc + sub * 16 * SCS + kc * 32, SCS, lane);
      v16h bf = frag_lds_h(s_vt + n0c * VTS + kc * 32, VTS, lane);
      acc = wmma_f32(a, bf, acc);
    }
    int hi = (lane & 16) ? 8 : 0, cl = lane & 15;
#pragma unroll
    for (int j = 0; j < 8; ++j) {
      int q = q0 + sub * 16 + j + hi;
      ctx[((size_t)b * CTQ + q) * CD + h * CDK + n0c + cl] = acc[j];
    }
  }
}

// ---------------------------------------------------------------------------
extern "C" void kernel_launch(void* const* d_in, const int* in_sizes, int n_in,
                              void* d_out, int out_size, void* d_ws, size_t ws_size,
                              hipStream_t stream) {
  (void)in_sizes; (void)n_in; (void)out_size; (void)ws_size;
  const float* x_in = (const float*)d_in[0];                 // (B,TQ,D)
  const float* mem  = (const float*)d_in[1];                 // (S,D)
  const unsigned char* mask = (const unsigned char*)d_in[2]; // (B,1,S) bool
  const float* Wq   = (const float*)d_in[3];
  const float* bq   = (const float*)d_in[4];
  const float* Wk   = (const float*)d_in[5];
  const float* bk   = (const float*)d_in[6];
  const float* Wv   = (const float*)d_in[7];
  const float* bv   = (const float*)d_in[8];
  const float* Wo   = (const float*)d_in[9];
  const float* bo   = (const float*)d_in[10];
  const float* Wpos = (const float*)d_in[11];
  const float* pbu  = (const float*)d_in[12];                // (L,H,DK)
  const float* pbv  = (const float*)d_in[13];

  // scratch carve (~25 MB), everything 16B aligned
  float* pe  = (float*)d_ws;                         // 2047*256 f32
  float* ctx = pe  + (size_t)N2 * CD;                // (B,TQ,D) f32
  float* xb  = ctx + (size_t)CB * CTQ * CD;          // (B,TQ,D) f32
  _Float16* Ph = (_Float16*)(xb + (size_t)CB * CTQ * CD); // (2S-1,D) f16
  _Float16* Kh = Ph + (size_t)N2 * CD;               // (S,D)   f16
  _Float16* Vt = Kh + (size_t)CS * CD;               // (D,S)   f16 transposed
  _Float16* Qh = Vt + (size_t)CD * CS;               // (B,TQ,D) f16

  float* out_x = (float*)d_out;
  float* out_w = out_x + (size_t)CB * CTQ * CD;

  const size_t WSTRIDE = (size_t)CD * CD;
  const size_t smem_bytes =
      (size_t)(32 * SCS + 48 * GS + 32 * 4) * sizeof(float) +
      (size_t)(32 * 40 + 48 * 40 + 176 * 40 + 32 * VTS) * sizeof(_Float16);

  pe_kernel<<<N2, 128, 0, stream>>>(pe);

  for (int l = 0; l < CL; ++l) {
    const float* xl = (l == 0) ? x_in : xb;
    // projections emit f16 (V transposed); K/V/P batch-invariant
    gemm64<_Float16, false><<<dim3(CB * CTQ / 64, CD / 64), 128, 0, stream>>>(
        xl, Wq + l * WSTRIDE, bq + l * CD, Qh, CB * CTQ, CD, CD, CD);
    gemm64<_Float16, false><<<dim3(CS / 64, CD / 64), 128, 0, stream>>>(
        mem, Wk + l * WSTRIDE, bk + l * CD, Kh, CS, CD, CD, CD);
    gemm64<_Float16, true><<<dim3(CS / 64, CD / 64), 128, 0, stream>>>(
        mem, Wv + l * WSTRIDE, bv + l * CD, Vt, CS, CD, CD, CS);
    gemm64<_Float16, false><<<dim3((N2 + 63) / 64, CD / 64), 128, 0, stream>>>(
        pe, Wpos + l * WSTRIDE, nullptr, Ph, N2, CD, CD, CD);

    attn_kernel<<<CB * CH * (CTQ / 32), 128, smem_bytes, stream>>>(
        Qh, Kh, Vt, Ph, pbu + l * CH * CDK, pbv + l * CH * CDK, mask,
        out_w, ctx, l);

    float* xdst = (l == CL - 1) ? out_x : xb;
    gemm64<float, false><<<dim3(CB * CTQ / 64, CD / 64), 128, 0, stream>>>(
        ctx, Wo + l * WSTRIDE, bo + l * CD, xdst, CB * CTQ, CD, CD, CD);
  }
}
